// Attention_17995912970857
// MI455X (gfx1250) — matrix operand order repaired and checked by reading, not compiled, not run on hardware
//
#include <hip/hip_runtime.h>
#include <hip/hip_bf16.h>
#include <math.h>

// ---------------------------------------------------------------------------
// MI455X (gfx1250) attention: RMSNorm -> QKV gemm -> rotary -> causal
// softcapped flash attention -> out gemm.  All matmuls bf16 WMMA
// (v_wmma_f32_16x16x32_bf16, f32 accum).  GEMMs and attention stage tiles
// in LDS via global_load_async_to_lds_b128 (ASYNCcnt), double-buffered.
// ---------------------------------------------------------------------------

#define B_    2
#define N_    2048
#define DIM_  1024
#define H_    16
#define DH_   64
#define ROWS_ (B_ * N_)          // 4096
#define QKVN_ (3 * H_ * DH_)     // 3072

typedef __attribute__((ext_vector_type(16))) __bf16 bf16x16;
typedef __attribute__((ext_vector_type(8)))  __bf16 bf16x8;
typedef __attribute__((ext_vector_type(8)))  float  f32x8;

union ABf {           // 16x32 A-frag or 32x16 B-frag (bf16), 8 VGPRs/lane
    bf16x16 v;
    bf16x8  h[2];
    __bf16  e[16];
};
union CF {            // 16x16 f32 C/D frag, 8 VGPRs/lane
    f32x8 v;
    float e[8];
};

static __device__ __forceinline__ f32x8 wmma_bf16(const ABf& a, const ABf& b, const f32x8& c) {
    return __builtin_amdgcn_wmma_f32_16x16x32_bf16(false, a.v, false, b.v,
                                                   (short)0, c, false, false);
}

// CDNA5 async global->LDS, 16B per lane, tracked by ASYNCcnt.
static __device__ __forceinline__ void async_ld16(const __bf16* lds_p, const __bf16* g_p) {
    unsigned lds_addr = (unsigned)(unsigned long long)(const void*)lds_p;
    asm volatile("global_load_async_to_lds_b128 %0, %1, off"
                 :: "v"(lds_addr), "v"((unsigned long long)(const void*)g_p)
                 : "memory");
}

// safe fast tanh: hardware-exp based, monotone, no NaN for large |z|
static __device__ __forceinline__ float fast_tanh(float z) {
    float a = __expf(-2.0f * fabsf(z));
    float t = (1.0f - a) / (1.0f + a);
    return copysignf(t, z);
}

// ---------------------------------------------------------------------------
// 1) RMSNorm: xn = x / ||x|| * sqrt(DIM) * (gamma+1), stored bf16
// ---------------------------------------------------------------------------
__global__ void k_rmsnorm(const float* __restrict__ x,
                          const float* __restrict__ gamma,
                          __bf16* __restrict__ xn) {
    const int row = blockIdx.x;                    // 0..4095
    const float* xr = x + (size_t)row * DIM_;
    float vals[4];
    float s = 0.f;
#pragma unroll
    for (int i = 0; i < 4; ++i) {
        vals[i] = xr[threadIdx.x + i * 256];
        s += vals[i] * vals[i];
    }
#pragma unroll
    for (int m = 16; m >= 1; m >>= 1) s += __shfl_xor(s, m, 32);
    __shared__ float red[8];
    if ((threadIdx.x & 31) == 0) red[threadIdx.x >> 5] = s;
    __syncthreads();
    float tot = 0.f;
#pragma unroll
    for (int w = 0; w < 8; ++w) tot += red[w];
    const float sc = 32.0f / fmaxf(sqrtf(tot), 1e-12f);   // sqrt(1024)=32
#pragma unroll
    for (int i = 0; i < 4; ++i) {
        const int c = threadIdx.x + i * 256;
        xn[(size_t)row * DIM_ + c] = (__bf16)(vals[i] * sc * (gamma[c] + 1.0f));
    }
}

// ---------------------------------------------------------------------------
// 2) Transpose + f32->bf16: wT[n*K + k] = w[k*Nn + n]
// ---------------------------------------------------------------------------
__global__ void k_transpose_bf16(const float* __restrict__ w,
                                 __bf16* __restrict__ wT, int K, int Nn) {
    const int idx = blockIdx.x * 256 + threadIdx.x;
    if (idx >= K * Nn) return;
    const int n = idx % Nn;
    const int k = idx / Nn;
    wT[(size_t)n * K + k] = (__bf16)w[idx];
}

// ---------------------------------------------------------------------------
// Shared GEMM core: C[128,128] tile, 8 waves each 32(M)x64(N), staged K=64.
// A (row-major [M,1024]) and B (col-major wT [N,1024]) tiles in LDS via
// async b128, double buffered; 16 WMMAs per wave per barrier pair.
// smem per buffer: A tile 128x64 then B tile 128x64 (bf16) = 32KB.
// ---------------------------------------------------------------------------
#define BK_ 64
#define TILE_ELTS (128 * BK_)    // 8192 elems = 16KB

static __device__ __forceinline__ void gemm_issue_tile(__bf16* buf,
                                                       const __bf16* gA, const __bf16* gB,
                                                       int mblk, int nblk, int k0) {
    const int c0 = threadIdx.x * 4;
#pragma unroll
    for (int u = 0; u < 4; ++u) {
        const int c   = c0 + u;         // 0..1023
        const int row = c >> 3;         // 0..127
        const int k8  = (c & 7) * 8;    // 0..56
        async_ld16(buf + row * BK_ + k8,
                   gA + (size_t)(mblk + row) * DIM_ + k0 + k8);
        async_ld16(buf + TILE_ELTS + row * BK_ + k8,
                   gB + (size_t)(nblk + row) * DIM_ + k0 + k8);
    }
}

static __device__ __forceinline__ void gemm_core(const __bf16* gA, const __bf16* gB,
                                                 __bf16* smem,       // [2][2*TILE_ELTS]
                                                 int mblk, int nblk,
                                                 CF acc[2][4]) {
    const int lane = threadIdx.x & 31;
    const int wave = threadIdx.x >> 5;
    const int wm = wave & 3;            // M chunk (32 rows)
    const int wn = wave >> 2;           // N chunk (64 cols)
    const int ln  = lane & 15;
    const int hlf = lane >> 4;

#pragma unroll
    for (int am = 0; am < 2; ++am)
#pragma unroll
        for (int nc = 0; nc < 4; ++nc)
#pragma unroll
            for (int r = 0; r < 8; ++r) acc[am][nc].e[r] = 0.f;

    gemm_issue_tile(smem, gA, gB, mblk, nblk, 0);            // prologue: buf 0

    for (int k0 = 0; k0 < DIM_; k0 += BK_) {
        const int cur = (k0 / BK_) & 1;
        const bool has_next = (k0 + BK_) < DIM_;
        if (has_next) {
            gemm_issue_tile(smem + (cur ^ 1) * (2 * TILE_ELTS), gA, gB,
                            mblk, nblk, k0 + BK_);
            asm volatile("s_wait_asynccnt 0x8" ::: "memory"); // cur buffer done
        } else {
            asm volatile("s_wait_asynccnt 0x0" ::: "memory");
        }
        __syncthreads();                                      // all waves' DMAs visible

        const __bf16* aB = smem + cur * (2 * TILE_ELTS);
        const __bf16* bB = aB + TILE_ELTS;
#pragma unroll
        for (int sub = 0; sub < 2; ++sub) {                   // two 32-deep K steps
            ABf a[2], b[4];
#pragma unroll
            for (int am = 0; am < 2; ++am) {
                const __bf16* p = aB + (wm * 32 + am * 16 + ln) * BK_ + sub * 32 + hlf * 16;
                a[am].h[0] = *(const bf16x8*)(p);
                a[am].h[1] = *(const bf16x8*)(p + 8);
            }
#pragma unroll
            for (int nc = 0; nc < 4; ++nc)
                b[nc].v = *(const bf16x16*)(bB + (wn * 64 + nc * 16 + ln) * BK_ +
                                            sub * 32 + hlf * 16);
#pragma unroll
            for (int am = 0; am < 2; ++am)
#pragma unroll
                for (int nc = 0; nc < 4; ++nc)
                    acc[am][nc].v = wmma_bf16(a[am], b[nc], acc[am][nc].v);
        }
        __syncthreads();                                      // reads done before refill
    }
}

// ---------------------------------------------------------------------------
// 3) QKV GEMM: [4096,1024]bf16 @ [1024,3072] -> scatter q,k [b,h,n,d], vT [b,h,d,n]
// ---------------------------------------------------------------------------
__global__ void __launch_bounds__(256) k_qkv_gemm(const __bf16* __restrict__ xn,
                                                  const __bf16* __restrict__ wT,
                                                  __bf16* __restrict__ qo,
                                                  __bf16* __restrict__ ko,
                                                  __bf16* __restrict__ vT) {
    __shared__ __attribute__((aligned(16))) __bf16 smem[2][2 * TILE_ELTS]; // 64 KB
    const int lane = threadIdx.x & 31;
    const int wave = threadIdx.x >> 5;
    const int wm = wave & 3, wn = wave >> 2;
    const int ln = lane & 15, hlf = lane >> 4;
    const int mblk = blockIdx.x * 128;
    const int nblk = blockIdx.y * 128;

    CF acc[2][4];
    gemm_core(xn, wT, &smem[0][0], mblk, nblk, acc);

#pragma unroll
    for (int am = 0; am < 2; ++am)
#pragma unroll
        for (int nc = 0; nc < 4; ++nc) {
            const int n0  = nblk + wn * 64 + nc * 16;   // same (sel,h) per chunk
            const int sel = n0 >> 10;                   // 0=q 1=k 2=v
            const int h   = (n0 >> 6) & 15;
            const int d   = (n0 & 63) + ln;
#pragma unroll
            for (int r = 0; r < 8; ++r) {
                const int row = mblk + wm * 32 + am * 16 + r + (hlf << 3);
                const int b = row >> 11, t = row & (N_ - 1);
                const size_t bh = (size_t)(b * H_ + h);
                const __bf16 val = (__bf16)acc[am][nc].e[r];
                if (sel == 0)      qo[(bh * N_ + t) * DH_ + d] = val;
                else if (sel == 1) ko[(bh * N_ + t) * DH_ + d] = val;
                else               vT[(bh * DH_ + d) * N_ + t] = val;  // transposed
            }
        }
}

// ---------------------------------------------------------------------------
// 4) Rotary on q,k (interleaved pairs), plus q *= DH^-0.5
// ---------------------------------------------------------------------------
__global__ void k_rotary(__bf16* __restrict__ q, __bf16* __restrict__ k,
                         const float* __restrict__ rot) {
    const int gid  = blockIdx.x * 256 + threadIdx.x;   // 65536 rows * 32 pairs
    const int pair = gid & 31;
    const int row  = gid >> 5;                         // [0, B*H*N)
    const int n    = row & (N_ - 1);
    const int d0   = pair * 2;
    const float f0 = rot[n * DH_ + d0];
    const float f1 = rot[n * DH_ + d0 + 1];
    const float c0 = __cosf(f0), s0 = __sinf(f0);
    const float c1 = __cosf(f1), s1 = __sinf(f1);
    const size_t base = (size_t)row * DH_ + d0;
    const float q0 = (float)q[base], q1 = (float)q[base + 1];
    const float k0 = (float)k[base], k1 = (float)k[base + 1];
    q[base]     = (__bf16)((q0 * c0 - q1 * s0) * 0.125f);
    q[base + 1] = (__bf16)((q1 * c1 + q0 * s1) * 0.125f);
    k[base]     = (__bf16)(k0 * c0 - k1 * s0);
    k[base + 1] = (__bf16)(k1 * c1 + k0 * s1);
}

// ---------------------------------------------------------------------------
// 5) Causal softcapped flash attention.  Block = 128 q rows (8 waves x 16).
//    K/V tiles for each 32-key block staged once per block in LDS via async
//    b128 (double buffered); all waves loop to the block causal bound and
//    skip compute (wave-uniform) past their own bound.
// ---------------------------------------------------------------------------
__global__ void __launch_bounds__(256) k_attn(const __bf16* __restrict__ q,
                                              const __bf16* __restrict__ kk,
                                              const __bf16* __restrict__ vT,
                                              __bf16* __restrict__ ao) {
    // per buffer: K tile [32 tok][64 d] (4KB) then V tile [64 d][32 tok] (4KB)
    __shared__ __attribute__((aligned(16))) __bf16 kvbuf[2][2 * 32 * 64];  // 16 KB
    __shared__ __bf16 ldsT[8][16 * 32];                                    // 8 KB
    const int lane  = threadIdx.x & 31;
    const int wave  = threadIdx.x >> 5;
    const int bh    = blockIdx.y;                        // 0..31
    const int qblock = blockIdx.x * 128;
    const int qbase  = qblock + wave * 16;               // 0..2047
    const size_t base = (size_t)bh * N_ * DH_;
    const int ln  = lane & 15;
    const int hlf = lane >> 4;
    const int koff = hlf * 16;
    const int tid = threadIdx.x;

    // Q fragments: 16x64 = two 16x32 A-frags
    ABf qa[2];
    const int qrow = qbase + ln;
#pragma unroll
    for (int s = 0; s < 2; ++s) {
        qa[s].h[0] = *(const bf16x8*)(q + base + (size_t)qrow * DH_ + s * 32 + koff);
        qa[s].h[1] = *(const bf16x8*)(q + base + (size_t)qrow * DH_ + s * 32 + koff + 8);
    }

    CF o[4];
    float mi[8], li[8];
#pragma unroll
    for (int nc = 0; nc < 4; ++nc)
#pragma unroll
        for (int r = 0; r < 8; ++r) o[nc].e[r] = 0.f;
#pragma unroll
    for (int r = 0; r < 8; ++r) { mi[r] = -1e30f; li[r] = 0.f; }

    const int my_nkb  = (qbase + 16 + 31) >> 5;          // this wave's causal bound
    const int blk_nkb = qblock / 32 + 4;                 // block causal bound

    // per-thread DMA shares (one K chunk + one V chunk per tile)
    const int ktok = tid >> 3, kd8 = (tid & 7) * 8;      // K: [32 tok][64 d]
    const int vd   = tid >> 2, vt8 = (tid & 3) * 8;      // V: [64 d][32 tok]

    // prologue: stage tile 0 into buffer 0
    async_ld16(&kvbuf[0][ktok * 64 + kd8],        kk + base + (size_t)ktok * DH_ + kd8);
    async_ld16(&kvbuf[0][2048 + vd * 32 + vt8],   vT + base + (size_t)vd * N_ + vt8);

    for (int jb = 0; jb < blk_nkb; ++jb) {
        const int cur = jb & 1;
        const int j0  = jb * 32;
        if (jb + 1 < blk_nkb) {
            const int jn = j0 + 32;
            async_ld16(&kvbuf[cur ^ 1][ktok * 64 + kd8],
                       kk + base + (size_t)(jn + ktok) * DH_ + kd8);
            async_ld16(&kvbuf[cur ^ 1][2048 + vd * 32 + vt8],
                       vT + base + (size_t)vd * N_ + jn + vt8);
            asm volatile("s_wait_asynccnt 0x2" ::: "memory");
        } else {
            asm volatile("s_wait_asynccnt 0x0" ::: "memory");
        }
        __syncthreads();                                  // all DMAs for cur visible

        if (jb < my_nkb) {                                // wave-uniform skip
            const __bf16* kt = kvbuf[cur];
            const __bf16* vt = kvbuf[cur] + 2048;
            CF s0, s1;
#pragma unroll
            for (int r = 0; r < 8; ++r) { s0.e[r] = 0.f; s1.e[r] = 0.f; }
#pragma unroll
            for (int s = 0; s < 2; ++s) {                 // K step over d
                ABf b0, b1;
                b0.v = *(const bf16x16*)(kt + (ln)      * 64 + s * 32 + hlf * 16);
                b1.v = *(const bf16x16*)(kt + (16 + ln) * 64 + s * 32 + hlf * 16);
                s0.v = wmma_bf16(qa[s], b0, s0.v);
                s1.v = wmma_bf16(qa[s], b1, s1.v);
            }
            // softcap + causal mask + online softmax (row = qbase + r + 8*hlf)
#pragma unroll
            for (int r = 0; r < 8; ++r) {
                const int row = qbase + r + (hlf << 3);
                float a0 = 50.f * fast_tanh(s0.e[r] * 0.02f);
                float a1 = 50.f * fast_tanh(s1.e[r] * 0.02f);
                if (j0 + ln      > row) a0 = -1e30f;
                if (j0 + 16 + ln > row) a1 = -1e30f;
                float mx = fmaxf(a0, a1);
#pragma unroll
                for (int off = 8; off >= 1; off >>= 1) mx = fmaxf(mx, __shfl_xor(mx, off, 16));
                const float mnew  = fmaxf(mi[r], mx);
                const float alpha = __expf(mi[r] - mnew);
                const float p0 = __expf(a0 - mnew);
                const float p1 = __expf(a1 - mnew);
                float ls = p0 + p1;
#pragma unroll
                for (int off = 8; off >= 1; off >>= 1) ls += __shfl_xor(ls, off, 16);
                li[r] = li[r] * alpha + ls;
                mi[r] = mnew;
                s0.e[r] = p0; s1.e[r] = p1;
                o[0].e[r] *= alpha; o[1].e[r] *= alpha;
                o[2].e[r] *= alpha; o[3].e[r] *= alpha;
            }
            // C-frag -> A-frag relayout through wave-private LDS tile
            __bf16* tile = ldsT[wave];
#pragma unroll
            for (int r = 0; r < 8; ++r) {
                const int rl = r + (hlf << 3);
                tile[rl * 32 + ln]      = (__bf16)s0.e[r];
                tile[rl * 32 + 16 + ln] = (__bf16)s1.e[r];
            }
            asm volatile("s_wait_dscnt 0x0" ::: "memory"); // RAW on wave-private LDS
            ABf pa;
            pa.h[0] = *(const bf16x8*)(tile + ln * 32 + koff);
            pa.h[1] = *(const bf16x8*)(tile + ln * 32 + koff + 8);
            // PV: V tile [d][tok] -> each B-frag one contiguous 32B LDS load
#pragma unroll
            for (int nc = 0; nc < 4; ++nc) {
                ABf bv;
                bv.v = *(const bf16x16*)(vt + (nc * 16 + ln) * 32 + hlf * 16);
                o[nc].v = wmma_bf16(pa, bv, o[nc].v);
            }
        }
        __syncthreads();                                  // reads done before refill
    }
    // normalize + store to ao[b, t, h*64+d] bf16
    const int b = bh >> 4, h = bh & 15;
#pragma unroll
    for (int nc = 0; nc < 4; ++nc)
#pragma unroll
        for (int r = 0; r < 8; ++r) {
            const int row = qbase + r + (hlf << 3);
            const int d = nc * 16 + ln;
            ao[((size_t)b * N_ + row) * (H_ * DH_) + h * DH_ + d] =
                (__bf16)(o[nc].e[r] / li[r]);
        }
}

// ---------------------------------------------------------------------------
// 6) Output projection: [4096,1024]bf16 @ [1024,1024] -> f32 out
// ---------------------------------------------------------------------------
__global__ void __launch_bounds__(256) k_out_gemm(const __bf16* __restrict__ ao,
                                                  const __bf16* __restrict__ wT,
                                                  float* __restrict__ out) {
    __shared__ __attribute__((aligned(16))) __bf16 smem[2][2 * TILE_ELTS]; // 64 KB
    const int lane = threadIdx.x & 31;
    const int wave = threadIdx.x >> 5;
    const int wm = wave & 3, wn = wave >> 2;
    const int ln = lane & 15, hlf = lane >> 4;
    const int mblk = blockIdx.x * 128;
    const int nblk = blockIdx.y * 128;

    CF acc[2][4];
    gemm_core(ao, wT, &smem[0][0], mblk, nblk, acc);

#pragma unroll
    for (int am = 0; am < 2; ++am)
#pragma unroll
        for (int nc = 0; nc < 4; ++nc)
#pragma unroll
            for (int r = 0; r < 8; ++r) {
                const int n   = nblk + wn * 64 + nc * 16 + ln;
                const int row = mblk + wm * 32 + am * 16 + r + (hlf << 3);
                out[(size_t)row * DIM_ + n] = acc[am][nc].e[r];
            }
}

// ---------------------------------------------------------------------------
extern "C" void kernel_launch(void* const* d_in, const int* in_sizes, int n_in,
                              void* d_out, int out_size, void* d_ws, size_t ws_size,
                              hipStream_t stream) {
    (void)in_sizes; (void)n_in; (void)out_size; (void)ws_size;
    const float* x     = (const float*)d_in[0];
    // d_in[1] = attn_mask (causal tril) -- statically known, unused
    const float* rot   = (const float*)d_in[2];
    const float* gamma = (const float*)d_in[3];
    const float* w_qkv = (const float*)d_in[4];
    const float* w_out = (const float*)d_in[5];
    float* out = (float*)d_out;

    char* ws = (char*)d_ws;
    size_t off = 0;
    __bf16* xn    = (__bf16*)(ws + off); off += (size_t)ROWS_ * DIM_  * 2;   // 8 MB
    __bf16* wqkvT = (__bf16*)(ws + off); off += (size_t)QKVN_ * DIM_  * 2;   // 6 MB
    __bf16* woutT = (__bf16*)(ws + off); off += (size_t)DIM_  * DIM_  * 2;   // 2 MB
    __bf16* qb    = (__bf16*)(ws + off); off += (size_t)B_*H_*N_*DH_  * 2;   // 8 MB
    __bf16* kb    = (__bf16*)(ws + off); off += (size_t)B_*H_*N_*DH_  * 2;   // 8 MB
    __bf16* vTb   = (__bf16*)(ws + off); off += (size_t)B_*H_*N_*DH_  * 2;   // 8 MB
    __bf16* aob   = (__bf16*)(ws + off); off += (size_t)ROWS_ * DIM_  * 2;   // 8 MB

    k_rmsnorm<<<ROWS_, 256, 0, stream>>>(x, gamma, xn);
    k_transpose_bf16<<<(DIM_ * QKVN_) / 256, 256, 0, stream>>>(w_qkv, wqkvT, DIM_, QKVN_);
    k_transpose_bf16<<<(DIM_ * DIM_) / 256, 256, 0, stream>>>(w_out, woutT, DIM_, DIM_);
    k_qkv_gemm<<<dim3(ROWS_ / 128, QKVN_ / 128), 256, 0, stream>>>(xn, wqkvT, qb, kb, vTb);
    k_rotary<<<(B_ * H_ * N_ * 32) / 256, 256, 0, stream>>>(qb, kb, rot);
    k_attn<<<dim3(N_ / 128, B_ * H_), 256, 0, stream>>>(qb, kb, vTb, aob);
    k_out_gemm<<<dim3(ROWS_ / 128, DIM_ / 128), 256, 0, stream>>>(aob, woutT, out);
}
